// AttModule_66468913873532
// MI455X (gfx1250) — matrix operand order repaired and checked by reading, not compiled, not run on hardware
//
#include <hip/hip_runtime.h>
#include <hip/hip_bf16.h>

// Model dims (from reference)
#define IN_N   50
#define OUT_N  10
#define FEAT   66
#define HID    256
#define VLEN   20
#define VSEQ   31
#define IFD    40
#define NBLK   12
#define NN     (FEAT*HID)   // 16896
#define PI_F   3.14159265358979f

typedef __attribute__((ext_vector_type(16))) _Float16 v16h;
typedef __attribute__((ext_vector_type(8)))  float    v8f;

__device__ __forceinline__ v8f vzero8() {
  v8f z;
#pragma unroll
  for (int i = 0; i < 8; ++i) z[i] = 0.f;
  return z;
}

__device__ __forceinline__ v8f wmma16(v16h a, v16h b, v8f c) {
  // D = A(16x32 f16) * B(32x16 f16) + C(16x16 f32)
  return __builtin_amdgcn_wmma_f32_16x16x32_f16(false, a, false, b, (short)0, c, false, false);
}

// A fragment (16x32 f16) from row-major source, leading dim ld.
// Lanes 0-15: M=lane, elems 0-15 = K 0..15
// Lanes 16-31: M=lane-16, elems 0-15 = K 16..31   (contiguous 32B per lane)
__device__ __forceinline__ v16h load_a_f16(const _Float16* base, int ld, int row0, int k0) {
  const int lane = threadIdx.x & 31;
  const int hi = lane >> 4;
  const _Float16* p = base + (size_t)(row0 + (lane & 15)) * ld + k0;
  v16h a;
#pragma unroll
  for (int j = 0; j < 8; ++j) {
    a[j]     = p[hi*16 + j];
    a[8 + j] = p[hi*16 + 8 + j];
  }
  return a;
}

// B fragment (32x16 f16) stored pre-swizzled: frag*512 + lane*16 + j halves,
// element j of lane <-> K = (lane>>4)*16 + j, N = lane&15.  One 32B load/lane.
__device__ __forceinline__ v16h load_frag(const _Float16* base, int frag) {
  const int lane = threadIdx.x & 31;
  return *(const v16h*)(base + ((size_t)frag * 32 + lane) * 16);
}

// frag-layout store address for element (k,n) of a (K x N16*16) B matrix
__device__ __forceinline__ int frag_addr(int k, int n, int ntiles) {
  return ((((k >> 5) * ntiles + (n >> 4)) * 32) + ((k >> 4) & 1) * 16 + (n & 15)) * 16 + (k & 15);
}

// ---------------------------------------------------------------------------
// Prep: convert / pad / swizzle all GEMM weights to f16 into workspace
// A-side (Wk1, Wk2, Aatt): row-major.  B-side (Wgin, Wblk, Wgout): frag order.
// ---------------------------------------------------------------------------
__global__ __launch_bounds__(256) void prep_weights(
    const float* __restrict__ k_w1, const float* __restrict__ k_w2,
    const float* __restrict__ gin_w, const float* __restrict__ blk_w,
    const float* __restrict__ gout_w, const float* __restrict__ gin_att,
    const float* __restrict__ blk_att, const float* __restrict__ gout_att,
    _Float16* __restrict__ Wk1, _Float16* __restrict__ Wk2,
    _Float16* __restrict__ Wgin, _Float16* __restrict__ Wblk,
    _Float16* __restrict__ Wgout, _Float16* __restrict__ Aatt) {
  const int S0 = 256*416;        // k conv1 A, K padded 396->416 (row-major)
  const int S1 = 256*1280;       // k conv2 A (row-major)
  const int S2 = 64*256;         // gin_w  B-frags (K 40->64, N 256)
  const int S3 = 12*256*256;     // block weights B-frags (256x256)
  const int S4 = 256*32;         // gout_w B-frags (K 256, N 32; output cols<20 used)
  const int S5 = 14*80*96;       // 14 att matrices padded 66x66 -> 80x96 (row-major A)
  const int total = S0+S1+S2+S3+S4+S5;
  for (int i = blockIdx.x*blockDim.x + threadIdx.x; i < total;
       i += gridDim.x*blockDim.x) {
    int j = i;
    if (j < S0) {
      int h = j / 416, k = j % 416;
      Wk1[j] = (_Float16)((k < 396) ? k_w1[h*396 + k] : 0.f);
    } else if ((j -= S0) < S1) {
      Wk2[j] = (_Float16)k_w2[j];
    } else if ((j -= S1) < S2) {
      // frag decode: j = frag*512 + lane*16 + e ; ntiles = 16
      int frag = j >> 9, r = j & 511;
      int lane = r >> 4, e = r & 15;
      int kt = frag >> 4, nt = frag & 15;
      int k = kt*32 + (lane >> 4)*16 + e;
      int n = nt*16 + (lane & 15);
      Wgin[j] = (_Float16)((k < 40) ? gin_w[k*256 + n] : 0.f);
    } else if ((j -= S2) < S3) {
      int blk = j >> 16, jj = j & 65535;       // 65536 per block, ntiles = 16
      int frag = jj >> 9, r = jj & 511;
      int lane = r >> 4, e = r & 15;
      int kt = frag >> 4, nt = frag & 15;
      int k = kt*32 + (lane >> 4)*16 + e;
      int n = nt*16 + (lane & 15);
      Wblk[j] = (_Float16)blk_w[blk*65536 + k*256 + n];
    } else if ((j -= S3) < S4) {
      int frag = j >> 9, r = j & 511;          // ntiles = 2
      int lane = r >> 4, e = r & 15;
      int kt = frag >> 1, nt = frag & 1;
      int k = kt*32 + (lane >> 4)*16 + e;
      int n = nt*16 + (lane & 15);
      Wgout[j] = (_Float16)gout_w[k*40 + n];
    } else {
      j -= S4;
      int a = j / 7680, rem = j % 7680;
      int m = rem / 96, k = rem % 96;
      float v = 0.f;
      if (m < 66 && k < 66) {
        if (a == 0)       v = gin_att[m*66 + k];
        else if (a <= 12) v = blk_att[(a-1)*4356 + m*66 + k];
        else              v = gout_att[m*66 + k];
      }
      Aatt[j] = (_Float16)v;
    }
  }
}

// ---------------------------------------------------------------------------
// Kernel 1: per-sample convs (WMMA im2col), attention weights, gcn_inp
// ---------------------------------------------------------------------------
__global__ __launch_bounds__(256) void attn_conv_kernel(
    const float* __restrict__ x,
    const float* __restrict__ q_w1, const float* __restrict__ q_b1,
    const float* __restrict__ q_w2, const float* __restrict__ q_b2,
    const _Float16* __restrict__ Wk1, const float* __restrict__ k_b1,
    const _Float16* __restrict__ Wk2, const float* __restrict__ k_b2,
    float* __restrict__ gin_buf) {
  __shared__ float    xs[IN_N*FEAT];                          // [pos][feat]
  __shared__ _Float16 o1[256*40];                             // conv1 out + pad
  __shared__ __attribute__((aligned(32))) _Float16 Pbuf[40960]; // im2col frags (union)
  __shared__ float    kf[256*32];                             // conv2 out (relu)
  __shared__ float    o1q[256*5];
  __shared__ float    qv[256];
  __shared__ float    dctm[400];   // dct[l][m] = w_l cos(pi(m+.5)l/20)
  __shared__ float    att_s[32];

  const int tid = threadIdx.x;
  const int b = blockIdx.x;
  const float* xb = x + (size_t)b * (IN_N*FEAT);

  for (int i = tid; i < IN_N*FEAT; i += 256) xs[i] = xb[i];
  for (int i = tid; i < 256*40; i += 256) o1[i] = (_Float16)0.f;
  for (int i = tid; i < 400; i += 256) {
    int l = i / 20, m = i % 20;
    float wl = (l == 0) ? 0.2236067977f : 0.3162277660f; // sqrt(1/20), sqrt(2/20)
    dctm[i] = wl * cosf(PI_F * (m + 0.5f) * l / 20.f);
  }
  __syncthreads();

  // Build conv1 im2col panel directly in B-fragment order (13 kt x 3 nt)
  for (int i = tid; i < 13*3*512; i += 256) {
    const int frag = i >> 9, r = i & 511;
    const int lne = r >> 4, e = r & 15;
    const int kt = frag / 3, nt = frag - kt*3;
    const int k = kt*32 + (lne >> 4)*16 + e;
    const int p = nt*16 + (lne & 15);
    float v = 0.f;
    if (k < 396 && p < 35) {
      const int c = k / 6, jj = k - c*6;   // k = c*6 + j
      v = xs[(p + jj)*FEAT + c];
    }
    Pbuf[i] = (_Float16)v;
  }
  __syncthreads();

  const int w  = tid >> 5;
  const int lane = tid & 31;
  const int ln = lane & 15;
  const int hi = lane >> 4;

  // ---- k conv1: M=256, N=48(35), K=416(396); wave: 2 mtiles x 3 ntiles
  {
    v8f acc[2][3];
#pragma unroll
    for (int a = 0; a < 2; ++a)
#pragma unroll
      for (int c = 0; c < 3; ++c) acc[a][c] = vzero8();
    for (int ks = 0; ks < 13; ++ks) {
      v16h b0 = load_frag(Pbuf, ks*3 + 0);
      v16h b1 = load_frag(Pbuf, ks*3 + 1);
      v16h b2 = load_frag(Pbuf, ks*3 + 2);
#pragma unroll
      for (int mi = 0; mi < 2; ++mi) {
        v16h a = load_a_f16(Wk1, 416, (w*2+mi)*16, ks*32);
        acc[mi][0] = wmma16(a, b0, acc[mi][0]);
        acc[mi][1] = wmma16(a, b1, acc[mi][1]);
        acc[mi][2] = wmma16(a, b2, acc[mi][2]);
      }
    }
#pragma unroll
    for (int mi = 0; mi < 2; ++mi)
#pragma unroll
      for (int nt = 0; nt < 3; ++nt)
#pragma unroll
        for (int r = 0; r < 8; ++r) {
          const int m = (w*2+mi)*16 + hi*8 + r;
          const int p = nt*16 + ln;
          if (p < 35) {
            float v = acc[mi][nt][r] + k_b1[m];
            o1[m*40 + p] = (_Float16)fmaxf(v, 0.f);
          }
        }
  }
  __syncthreads();

  // Build conv2 im2col panel in fragment order (40 kt x 2 nt); reuses Pbuf
  for (int i = tid; i < 40*2*512; i += 256) {
    const int frag = i >> 9, r = i & 511;
    const int lne = r >> 4, e = r & 15;
    const int kt = frag >> 1, nt = frag & 1;
    const int k = kt*32 + (lne >> 4)*16 + e;
    const int p = nt*16 + (lne & 15);
    const int c = k / 5, jj = k - c*5;       // k = c*5 + j ; p+jj <= 35 (zero pad)
    Pbuf[i] = o1[c*40 + p + jj];
  }
  __syncthreads();

  // ---- k conv2: M=256, N=32(31), K=1280; wave: 2 mtiles x 2 ntiles
  {
    v8f acc[2][2];
#pragma unroll
    for (int a = 0; a < 2; ++a)
#pragma unroll
      for (int c = 0; c < 2; ++c) acc[a][c] = vzero8();
    for (int ks = 0; ks < 40; ++ks) {
      v16h b0 = load_frag(Pbuf, ks*2 + 0);
      v16h b1 = load_frag(Pbuf, ks*2 + 1);
#pragma unroll
      for (int mi = 0; mi < 2; ++mi) {
        v16h a = load_a_f16(Wk2, 1280, (w*2+mi)*16, ks*32);
        acc[mi][0] = wmma16(a, b0, acc[mi][0]);
        acc[mi][1] = wmma16(a, b1, acc[mi][1]);
      }
    }
#pragma unroll
    for (int mi = 0; mi < 2; ++mi)
#pragma unroll
      for (int nt = 0; nt < 2; ++nt)
#pragma unroll
        for (int r = 0; r < 8; ++r) {
          const int m = (w*2+mi)*16 + hi*8 + r;
          const int p = nt*16 + ln;
          if (p < 31) kf[m*32 + p] = fmaxf(acc[mi][nt][r] + k_b2[m], 0.f);
        }
  }
  __syncthreads();

  // ---- q path (tiny N): plain f32 VALU
  for (int idx = tid; idx < 256*5; idx += 256) {
    const int h = idx / 5, p = idx % 5;
    float s = q_b1[h];
    const float* wr = q_w1 + (size_t)h*396;
    for (int c = 0; c < 66; ++c)
#pragma unroll
      for (int j = 0; j < 6; ++j)
        s += xs[(40 + p + j)*FEAT + c] * wr[c*6 + j];
    o1q[idx] = fmaxf(s, 0.f);
  }
  __syncthreads();
  {
    const int h = tid;
    float s = q_b2[h];
    const float* wr = q_w2 + (size_t)h*1280;
    for (int c = 0; c < 256; ++c)
#pragma unroll
      for (int j = 0; j < 5; ++j)
        s += o1q[c*5 + j] * wr[c*5 + j];
    qv[h] = fmaxf(s, 0.f);
  }
  __syncthreads();

  // ---- scores + normalization
  if (tid < 31) {
    float s = 0.f;
    for (int h = 0; h < 256; ++h) s += qv[h] * kf[h*32 + tid];
    att_s[tid] = s;
  }
  __syncthreads();
  if (tid == 0) {
    float t = 0.f;
    for (int s = 0; s < 31; ++s) t += att_s[s];
    const float inv = 1.f / t;
    for (int s = 0; s < 31; ++s) att_s[s] *= inv;
  }
  __syncthreads();

  // ---- gcn_inp = [padded_query | att_final]  (66 x 40)
  float* gb = gin_buf + (size_t)b * (FEAT*IFD);
  for (int idx = tid; idx < FEAT*VLEN; idx += 256) {
    const int f = idx / 20, m = idx % 20;
    float s = 0.f;
#pragma unroll
    for (int j = 0; j < 20; ++j) {
      const int pos = (j < 10) ? (40 + j) : 49;
      s += xs[pos*FEAT + f] * dctm[j*20 + m];
    }
    gb[f*40 + m] = s;
    // att_final with the reference's raw reshape: t=f*31+s -> (s'=t/66, f'=t%66)
    float s2 = 0.f;
    for (int sx = 0; sx < 31; ++sx) {
      const int t = f*31 + sx;
      const int sp = t / 66, fp = t - sp*66;
      float acc2 = 0.f;
#pragma unroll
      for (int lp = 0; lp < 20; ++lp)
        acc2 += xs[(sp + lp)*FEAT + fp] * dctm[lp*20 + m];
      s2 += att_s[sx] * acc2;
    }
    gb[f*40 + 20 + m] = s2;
  }
}

// ---------------------------------------------------------------------------
// Kernel 2: fused GCN trunk, one workgroup per sample, activations in LDS
// ---------------------------------------------------------------------------
// gemm1: Tf(frag) = Hf(66xK f16,row-major) @ Bw(KxN=256 f16, frag order)
__device__ __forceinline__ void gemm1(const _Float16* Hf, _Float16* Tf,
                                      const _Float16* __restrict__ Bw, int nk) {
  const int tid = threadIdx.x;
  const int w = tid >> 5, lane = tid & 31, ln = lane & 15, hi = lane >> 4;
  v8f acc[5][2];
#pragma unroll
  for (int a = 0; a < 5; ++a) { acc[a][0] = vzero8(); acc[a][1] = vzero8(); }
  for (int ks = 0; ks < nk; ++ks) {
    v16h b0 = load_frag(Bw, ks*16 + w*2);
    v16h b1 = load_frag(Bw, ks*16 + w*2 + 1);
#pragma unroll
    for (int mt = 0; mt < 5; ++mt) {
      v16h a = load_a_f16(Hf, 256, mt*16, ks*32);
      acc[mt][0] = wmma16(a, b0, acc[mt][0]);
      acc[mt][1] = wmma16(a, b1, acc[mt][1]);
    }
  }
#pragma unroll
  for (int mt = 0; mt < 5; ++mt)
#pragma unroll
    for (int nt = 0; nt < 2; ++nt)
#pragma unroll
      for (int r = 0; r < 8; ++r) {
        const int m = mt*16 + hi*8 + r;              // K index of gemm2
        if (m < 66) {
          const int n = w*32 + nt*16 + ln;
          Tf[frag_addr(m, n, 16)] = (_Float16)acc[mt][nt][r];
        }
      }
}

// gemm2: acc = Aa(66x66 pad 80x96 f16,row-major) @ Tf(frag), epilogue bias/bn/tanh.
// mode 0: write Hf only; mode 1: +=Yf residual, write Yf & Hf; mode 2: write Yf & Hf
__device__ __forceinline__ void gemm2(const _Float16* Tf, _Float16* Hf, float* Yf,
                                      const _Float16* __restrict__ Aa,
                                      const float* __restrict__ bias,
                                      const float* __restrict__ g,
                                      const float* __restrict__ be, int mode) {
  const int tid = threadIdx.x;
  const int w = tid >> 5, lane = tid & 31, ln = lane & 15, hi = lane >> 4;
  v8f acc[5][2];
#pragma unroll
  for (int a = 0; a < 5; ++a) { acc[a][0] = vzero8(); acc[a][1] = vzero8(); }
#pragma unroll
  for (int ks = 0; ks < 3; ++ks) {
    v16h b0 = load_frag(Tf, ks*16 + w*2);
    v16h b1 = load_frag(Tf, ks*16 + w*2 + 1);
#pragma unroll
    for (int mt = 0; mt < 5; ++mt) {
      v16h a = load_a_f16(Aa, 96, mt*16, ks*32);
      acc[mt][0] = wmma16(a, b0, acc[mt][0]);
      acc[mt][1] = wmma16(a, b1, acc[mt][1]);
    }
  }
  const float inv = 0.9999950000375f;  // 1/sqrt(1+1e-5)
#pragma unroll
  for (int mt = 0; mt < 5; ++mt)
#pragma unroll
    for (int nt = 0; nt < 2; ++nt)
#pragma unroll
      for (int r = 0; r < 8; ++r) {
        const int m = mt*16 + hi*8 + r;
        if (m >= 66) continue;
        const int n = w*32 + nt*16 + ln;
        const int idx = m*256 + n;
        float v = acc[mt][nt][r] + bias[n];
        v = tanhf(v * inv * g[idx] + be[idx]);
        if (mode == 1) v += Yf[idx];
        if (mode != 0) Yf[idx] = v;
        Hf[idx] = (_Float16)v;
      }
}

__global__ __launch_bounds__(256) void gcn_kernel(
    const float* __restrict__ gin_buf,
    const _Float16* __restrict__ Wgin, const _Float16* __restrict__ Wblk,
    const _Float16* __restrict__ Wgout, const _Float16* __restrict__ Aatt,
    const float* __restrict__ gin_b, const float* __restrict__ bn_in_g,
    const float* __restrict__ bn_in_b, const float* __restrict__ blk_b,
    const float* __restrict__ blk_bn_g, const float* __restrict__ blk_bn_b,
    const float* __restrict__ gout_b, float* __restrict__ out) {
  __shared__ _Float16 Hf[80*256];   // A operand (f16), rows 66..79 zero
  __shared__ __attribute__((aligned(32))) _Float16 Tf[48*512]; // staging, frag order
  __shared__ float    Yf[66*256];   // f32 residual-exact y
  __shared__ float    Gf[66*40];    // gcn_inp -> gcn_out
  __shared__ float    idctm[400];   // idct[l][m] = w_m cos(pi(l+.5)m/20)

  const int tid = threadIdx.x;
  const int b = blockIdx.x;
  const float* gb = gin_buf + (size_t)b * (FEAT*IFD);

  for (int i = tid; i < 80*256; i += 256) Hf[i] = (_Float16)0.f;
  for (int i = tid; i < 48*512; i += 256) Tf[i] = (_Float16)0.f;
  for (int i = tid; i < FEAT*IFD; i += 256) Gf[i] = gb[i];
  for (int i = tid; i < 400; i += 256) {
    int l = i / 20, m = i % 20;
    float wm = (m == 0) ? 0.2236067977f : 0.3162277660f;
    idctm[i] = wm * cosf(PI_F * (l + 0.5f) * m / 20.f);
  }
  __syncthreads();
  for (int i = tid; i < FEAT*IFD; i += 256)
    Hf[(i/40)*256 + (i%40)] = (_Float16)Gf[i];
  __syncthreads();

  // input gc: K=40 (padded 64)
  gemm1(Hf, Tf, Wgin, 2);
  __syncthreads();
  gemm2(Tf, Hf, Yf, Aatt, gin_b, bn_in_g, bn_in_b, 2);
  __syncthreads();

  // 12 blocks (6 residual pairs)
  for (int i = 0; i < NBLK; ++i) {
    if (i + 1 < NBLK)
      __builtin_prefetch(Wblk + (size_t)(i+1)*65536, 0, 3);
    gemm1(Hf, Tf, Wblk + (size_t)i*65536, 8);
    __syncthreads();
    gemm2(Tf, Hf, Yf, Aatt + (size_t)(1+i)*7680, blk_b + i*256,
          blk_bn_g + (size_t)i*NN, blk_bn_b + (size_t)i*NN, (i & 1) ? 1 : 0);
    __syncthreads();
  }

  // gout gc (N=32; only first 20 output cols feed the idct)
  const int w = tid >> 5, lane = tid & 31, ln = lane & 15, hi = lane >> 4;
  if (w < 5) {
    v8f acc[2] = {vzero8(), vzero8()};
    for (int ks = 0; ks < 8; ++ks) {
      v16h a  = load_a_f16(Hf, 256, w*16, ks*32);
      v16h b0 = load_frag(Wgout, ks*2 + 0);
      v16h b1 = load_frag(Wgout, ks*2 + 1);
      acc[0] = wmma16(a, b0, acc[0]);
      acc[1] = wmma16(a, b1, acc[1]);
    }
#pragma unroll
    for (int nt = 0; nt < 2; ++nt)
#pragma unroll
      for (int r = 0; r < 8; ++r) {
        const int m = w*16 + hi*8 + r;
        if (m < 66) Tf[frag_addr(m, nt*16 + ln, 16)] = (_Float16)acc[nt][r];
      }
  }
  __syncthreads();
  if (w < 5) {
    v8f acc[2] = {vzero8(), vzero8()};
#pragma unroll
    for (int ks = 0; ks < 3; ++ks) {
      v16h a  = load_a_f16(Aatt + (size_t)13*7680, 96, w*16, ks*32);
      v16h b0 = load_frag(Tf, ks*16 + 0);
      v16h b1 = load_frag(Tf, ks*16 + 1);
      acc[0] = wmma16(a, b0, acc[0]);
      acc[1] = wmma16(a, b1, acc[1]);
    }
#pragma unroll
    for (int nt = 0; nt < 2; ++nt)
#pragma unroll
      for (int r = 0; r < 8; ++r) {
        const int m = w*16 + hi*8 + r;
        const int n = nt*16 + ln;
        if (m < 66) {
          float v = acc[nt][r] + gout_b[n] + Gf[m*40 + n];
          Gf[m*40 + n] = v;   // gcn_out (cols 0..31; only 0..19 used below)
        }
      }
  }
  __syncthreads();

  // final idct: out[f][m] = sum_l gcn_out[f][l] * idct[l][m]
  float* ob = out + (size_t)b * (FEAT*VLEN);
  for (int i = tid; i < FEAT*VLEN; i += 256) {
    const int f = i / 20, m = i % 20;
    float s = 0.f;
#pragma unroll
    for (int l = 0; l < 20; ++l) s += Gf[f*40 + l] * idctm[l*20 + m];
    ob[i] = s;
  }
}

// ---------------------------------------------------------------------------
extern "C" void kernel_launch(void* const* d_in, const int* in_sizes, int n_in,
                              void* d_out, int out_size, void* d_ws, size_t ws_size,
                              hipStream_t stream) {
  const float* x        = (const float*)d_in[0];
  const float* q_w1     = (const float*)d_in[1];
  const float* q_b1     = (const float*)d_in[2];
  const float* q_w2     = (const float*)d_in[3];
  const float* q_b2     = (const float*)d_in[4];
  const float* k_w1     = (const float*)d_in[5];
  const float* k_b1     = (const float*)d_in[6];
  const float* k_w2     = (const float*)d_in[7];
  const float* k_b2     = (const float*)d_in[8];
  const float* gin_w    = (const float*)d_in[9];
  const float* gin_att  = (const float*)d_in[10];
  const float* gin_b    = (const float*)d_in[11];
  const float* bn_in_g  = (const float*)d_in[12];
  const float* bn_in_b  = (const float*)d_in[13];
  const float* blk_w    = (const float*)d_in[14];
  const float* blk_att  = (const float*)d_in[15];
  const float* blk_b    = (const float*)d_in[16];
  const float* blk_bn_g = (const float*)d_in[17];
  const float* blk_bn_b = (const float*)d_in[18];
  const float* gout_w   = (const float*)d_in[19];
  const float* gout_att = (const float*)d_in[20];
  const float* gout_b   = (const float*)d_in[21];

  const int B = in_sizes[0] / (IN_N * FEAT);

  char* ws = (char*)d_ws;
  size_t off = 0;
  auto alloc = [&](size_t bytes) -> void* {
    void* p = ws + off;
    off += (bytes + 255) & ~(size_t)255;
    return p;
  };
  _Float16* Wk1   = (_Float16*)alloc((size_t)256*416*2);
  _Float16* Wk2   = (_Float16*)alloc((size_t)256*1280*2);
  _Float16* Wgin  = (_Float16*)alloc((size_t)64*256*2);
  _Float16* Wblk  = (_Float16*)alloc((size_t)12*256*256*2);
  _Float16* Wgout = (_Float16*)alloc((size_t)256*32*2);
  _Float16* Aatt  = (_Float16*)alloc((size_t)14*80*96*2);
  float*    ginb  = (float*)alloc((size_t)B*FEAT*IFD*4);
  (void)ws_size; (void)n_in; (void)out_size;

  prep_weights<<<1024, 256, 0, stream>>>(k_w1, k_w2, gin_w, blk_w, gout_w,
                                         gin_att, blk_att, gout_att,
                                         Wk1, Wk2, Wgin, Wblk, Wgout, Aatt);
  attn_conv_kernel<<<B, 256, 0, stream>>>(x, q_w1, q_b1, q_w2, q_b2,
                                          Wk1, k_b1, Wk2, k_b2, ginb);
  gcn_kernel<<<B, 256, 0, stream>>>(ginb, Wgin, Wblk, Wgout, Aatt,
                                    gin_b, bn_in_g, bn_in_b,
                                    blk_b, blk_bn_g, blk_bn_b,
                                    gout_b, (float*)d_out);
}
